// Convalg_45715631898888
// MI455X (gfx1250) — 8-bit matrix operand order repaired and checked by reading, compiled, not run on hardware
//
#include <hip/hip_runtime.h>

typedef __attribute__((ext_vector_type(16))) int   v16i;
typedef __attribute__((ext_vector_type(8)))  float v8f;
typedef __attribute__((ext_vector_type(4)))  int   v4i;
typedef __attribute__((ext_vector_type(4)))  float v4f;

#define B_N     32
#define CIN     256
#define COUT    256
#define H_      56
#define W_      56
#define HP      58
#define KTOT    2304
#define KSTEPS  18

#define XQP_BYTES  (B_N * HP * HP * CIN)
#define BPK_BYTES  (KSTEPS * 16 * 32 * 64)
#define OFF_BPK    XQP_BYTES
#define OFF_ALPHA  (OFF_BPK + BPK_BYTES)
#define OFF_BIAS   (OFF_ALPHA + 1024)

__device__ __forceinline__ int sigma128(int kk) {
  return ((kk >> 4) & 1) * 64 + (kk >> 5) * 16 + (kk & 15);
}

__global__ __launch_bounds__(256) void quant_kernel(
    const float* __restrict__ x, unsigned char* __restrict__ xqp) {
  int yy = blockIdx.x % HP;
  int b  = blockIdx.x / HP;
  int c  = threadIdx.x;
  const int p = ((c >> 7) << 7) + sigma128(c & 127);
  unsigned char* row = xqp + (size_t)((b * HP + yy) * HP) * CIN;
  if (yy == 0 || yy == HP - 1) {
    for (int xx = 0; xx < HP; ++xx) row[xx * CIN + p] = 0;
    return;
  }
  row[0 * CIN + p]        = 0;
  row[(HP - 1) * CIN + p] = 0;
  const float* src = x + ((size_t)(b * CIN + c) * H_ + (yy - 1)) * W_;
  #pragma unroll
  for (int q4 = 0; q4 < 14; ++q4) {
    v4f v = *(const v4f*)(src + q4 * 4);
    #pragma unroll
    for (int j = 0; j < 4; ++j) {
      float f = v[j];
      row[(q4 * 4 + j + 1) * CIN + p] =
          f > 0.f ? 0x38 : (f < 0.f ? 0xB8 : 0x00);
    }
  }
}

__global__ __launch_bounds__(256) void wprep_kernel(
    const float* __restrict__ W, const float* __restrict__ gamma,
    const float* __restrict__ beta, const float* __restrict__ rmean,
    const float* __restrict__ rvar, unsigned char* __restrict__ Bpk,
    float* __restrict__ alpha, float* __restrict__ bias) {
  __shared__ float red[256];
  const int co = blockIdx.x, t = threadIdx.x;
  const float* wf = W + (size_t)co * KTOT;

  float s = 0.f;
  for (int j = t; j < KTOT; j += 256) s += wf[j];
  red[t] = s; __syncthreads();
  for (int o = 128; o > 0; o >>= 1) { if (t < o) red[t] += red[t + o]; __syncthreads(); }
  const float mean = red[0] / (float)KTOT;
  __syncthreads();

  float s2 = 0.f;
  for (int j = t; j < KTOT; j += 256) { float d = wf[j] - mean; s2 += d * d; }
  red[t] = s2; __syncthreads();
  for (int o = 128; o > 0; o >>= 1) { if (t < o) red[t] += red[t + o]; __syncthreads(); }
  const float stdv = sqrtf(red[0] / (float)(KTOT - 1));
  __syncthreads();

  float sa = 0.f;
  for (int j = t; j < KTOT; j += 256) sa += fabsf(wf[j] - mean);
  red[t] = sa; __syncthreads();
  for (int o = 128; o > 0; o >>= 1) { if (t < o) red[t] += red[t + o]; __syncthreads(); }
  const float meanabs = red[0] / (stdv * (float)KTOT);
  const float sw = exp2f(roundf(log2f(meanabs)));

  if (t == 0) {
    float inv = gamma[co] / sqrtf(rvar[co] + 1e-5f);
    alpha[co] = sw * inv;
    bias[co]  = beta[co] - rmean[co] * inv;
  }

  const int ntile = co >> 4, ncol = co & 15;
  for (int k = t; k < KTOT; k += 256) {
    int rs = k >> 8, ci = k & 255;
    float d = wf[ci * 9 + rs] - mean;
    unsigned char byte = d > 0.f ? 0x38 : (d < 0.f ? 0xB8 : 0x00);
    int kstep = k >> 7, kk = k & 127;
    int g = kk >> 5, w32 = kk & 31;
    int half = w32 >> 4, rem = w32 & 15;
    int vg = g * 4 + (rem >> 2), byt = rem & 3;
    int lane = ncol + (half << 4);
    Bpk[(((size_t)(kstep * 16 + ntile) * 32 + lane) << 6) + (vg << 2) + byt] = byte;
  }
}

__global__ __launch_bounds__(256) void bconv_kernel(
    const unsigned char* __restrict__ xqp, const unsigned char* __restrict__ Bpk,
    const float* __restrict__ alpha, const float* __restrict__ bias,
    float* __restrict__ out) {
  const int y  = blockIdx.x % H_;
  const int b  = blockIdx.x / H_;
  const int t  = threadIdx.x;
  const int lane = t & 31, wv = t >> 5;
  const int m  = lane & 15, hf = lane >> 4;

  v8f acc[4][2] = {};

  #pragma unroll 2
  for (int s = 0; s < KSTEPS; ++s) {
    const int rs = s >> 1;
    const int dy = rs / 3 - 1, dx = rs % 3 - 1, ci0 = (s & 1) << 7;

    union BF { v4i q[4]; v16i v; } Bf[2];
    const unsigned char* bb =
        Bpk + ((size_t)(s * 16 + wv * 2) << 11) + lane * 64;
    #pragma unroll
    for (int nt = 0; nt < 2; ++nt) {
      const v4i* bp = (const v4i*)(bb + nt * 2048);
      Bf[nt].q[0] = bp[0]; Bf[nt].q[1] = bp[1];
      Bf[nt].q[2] = bp[2]; Bf[nt].q[3] = bp[3];
    }

    const unsigned char* arow = xqp +
        (((size_t)(b * HP + (y + dy + 1)) * HP + (m + dx + 1)) << 8) +
        ci0 + hf * 64;

    union AF { v4i q[4]; v16i v; } Af[4];
    #pragma unroll
    for (int mt = 0; mt < 4; ++mt) {
      int w0 = mt * 16; if (mt == 3) w0 = 40;
      const v4i* ap = (const v4i*)(arow + ((size_t)w0 << 8));
      Af[mt].q[0] = ap[0]; Af[mt].q[1] = ap[1];
      Af[mt].q[2] = ap[2]; Af[mt].q[3] = ap[3];
    }

    #pragma unroll
    for (int mt = 0; mt < 4; ++mt) {
      acc[mt][0] = __builtin_amdgcn_wmma_f32_16x16x128_fp8_fp8(
          Af[mt].v, Bf[0].v, (short)0, acc[mt][0], false, false);
      acc[mt][1] = __builtin_amdgcn_wmma_f32_16x16x128_fp8_fp8(
          Af[mt].v, Bf[1].v, (short)0, acc[mt][1], false, false);
    }
  }

  const int ncol = lane & 15;
  #pragma unroll
  for (int nt = 0; nt < 2; ++nt) {
    const int cout = (wv * 2 + nt) * 16 + ncol;
    const float al = alpha[cout], bi = bias[cout];
    float* obase = out + ((size_t)(b * COUT + cout) * H_ + y) * W_;
    #pragma unroll
    for (int mt = 0; mt < 4; ++mt) {
      int w0 = mt * 16; if (mt == 3) w0 = 40;
      float* op = obase + w0 + hf * 8;
      v4f o0, o1;
      #pragma unroll
      for (int r = 0; r < 4; ++r) {
        float v0 = al * acc[mt][nt][r]     + bi;
        float v1 = al * acc[mt][nt][4 + r] + bi;
        o0[r] = fminf(1.f, fmaxf(-1.f, v0));
        o1[r] = fminf(1.f, fmaxf(-1.f, v1));
      }
      __builtin_nontemporal_store(o0, (v4f*)op);
      __builtin_nontemporal_store(o1, (v4f*)(op + 4));
    }
  }
}

extern "C" void kernel_launch(void* const* d_in, const int* in_sizes, int n_in,
                              void* d_out, int out_size, void* d_ws, size_t ws_size,
                              hipStream_t stream) {
  const float* x     = (const float*)d_in[0];
  const float* w     = (const float*)d_in[1];
  const float* gamma = (const float*)d_in[2];
  const float* beta  = (const float*)d_in[3];
  const float* rmean = (const float*)d_in[4];
  const float* rvar  = (const float*)d_in[5];
  float* out = (float*)d_out;

  unsigned char* ws   = (unsigned char*)d_ws;
  unsigned char* xqp  = ws;
  unsigned char* bpk  = ws + OFF_BPK;
  float* alpha        = (float*)(ws + OFF_ALPHA);
  float* bias         = (float*)(ws + OFF_BIAS);

  quant_kernel<<<B_N * HP, 256, 0, stream>>>(x, xqp);
  wprep_kernel<<<COUT, 256, 0, stream>>>(w, gamma, beta, rmean, rvar, bpk, alpha, bias);
  bconv_kernel<<<B_N * H_, 256, 0, stream>>>(xqp, bpk, alpha, bias, out);
}
